// SchNet_39084202393768
// MI455X (gfx1250) — matrix operand order repaired and checked by reading, not compiled, not run on hardware
//
#include <hip/hip_runtime.h>

#define HD    128          // hidden channels
#define RB    64           // num rbf
#define NATOM 40000
#define NEDGE 640000
#define NBLK  3
#define CUTF  5.0f

typedef __attribute__((ext_vector_type(16))) _Float16 v16h;
typedef __attribute__((ext_vector_type(8)))  _Float16 v8h;
typedef __attribute__((ext_vector_type(4)))  _Float16 v4h;
typedef __attribute__((ext_vector_type(8)))  float    v8f;

// ---------------------------------------------------------------- helpers

__device__ __forceinline__ v8f wmma16(v16h a, v16h b, v8f c) {
  // D = A(16x32 f16) * B(32x16 f16) + C(16x16 f32)
  return __builtin_amdgcn_wmma_f32_16x16x32_f16(
      /*neg_a=*/false, a, /*neg_b=*/false, b,
      /*c_mod=*/(short)0, c, /*reuse_a=*/false, /*reuse_b=*/false);
}

__device__ __forceinline__ v8f splat8(float v) {
  v8f r;
#pragma unroll
  for (int i = 0; i < 8; ++i) r[i] = v;
  return r;
}

#define LOG2E 1.4426950408889634f

// exp(y) with a single v_exp_f32 (args here are bounded, no range fixup needed)
__device__ __forceinline__ float fast_exp(float y) {
  return __builtin_amdgcn_exp2f(y * LOG2E);
}

// tanh(x) = 1 - 2/(exp2(2*log2e*x)+1): mul + v_exp + add + v_rcp + fma
// saturates correctly: x->+inf => 1, x->-inf => -1
__device__ __forceinline__ float fast_tanh(float x) {
  float e = __builtin_amdgcn_exp2f(x * (2.0f * LOG2E));
  return 1.0f - 2.0f * __builtin_amdgcn_rcpf(e + 1.0f);
}

// A-fragment (16x32 f16, ISA 7.12.2) = two contiguous 8-half runs per lane:
//   elements 0..7  : K = kbase + 8*hs      .. +8
//   elements 8..15 : K = kbase + 16 + 8*hs .. +8
__device__ __forceinline__ v16h load_a_lds(const _Float16* t, int kbase, int lane) {
  int row = lane & 15, hs = lane >> 4;
  const _Float16* p = t + row * HD + kbase + hs * 8;
  v8h lo = *(const v8h*)p;           // ds_load_b128
  v8h hi = *(const v8h*)(p + 16);    // ds_load_b128
  return __builtin_shufflevector(lo, hi, 0, 1, 2, 3, 4, 5, 6, 7,
                                         8, 9, 10, 11, 12, 13, 14, 15);
}

// A-fragment from fp32 row-major global memory (convert to f16)
__device__ __forceinline__ v16h load_a_f32(const float* p0, int kbase, int lane) {
  int hs = lane >> 4;
  const float* p = p0 + kbase + hs * 8;
  v8f lo = *(const v8f*)p;           // 2x global_load_b128
  v8f hi = *(const v8f*)(p + 16);    // 2x global_load_b128
  v16h a;
#pragma unroll
  for (int i = 0; i < 8; ++i) { a[i] = (_Float16)lo[i]; a[8 + i] = (_Float16)hi[i]; }
  return a;
}

// B-fragment from pre-packed weights: one contiguous 32B load per lane
__device__ __forceinline__ v16h load_b_packed(const _Float16* p, int kc, int ct, int lane) {
  return *(const v16h*)(p + (((kc * 8 + ct) * 32) + lane) * 16);  // 2x global_load_b128
}

// ---------------------------------------------------------------- kernels

// Pack a stacked weight (NBLK, K, HD) fp32 into f16 B-fragment-major layout:
// dst[blk][kc][ct][lane][e] = w[blk][(kc*32 + hs*8 + (e<8 ? e : e+8))][ct*16 + col]
__global__ void pack_b_kernel(const float* __restrict__ src,
                              _Float16* __restrict__ dst,
                              int kchunks, int total) {
  int i = blockIdx.x * blockDim.x + threadIdx.x;
  if (i >= total) return;
  int e    = i & 15;
  int lane = (i >> 4) & 31;
  int ct   = (i >> 9) & 7;
  int kc   = (i >> 12) % kchunks;
  int blk  = i / (kchunks * 4096);
  int K    = kchunks * 32;
  int col  = lane & 15, hs = lane >> 4;
  float v = src[(size_t)blk * K * HD + (kc * 32 + hs * 8 + ((e < 8) ? e : (e + 8))) * HD + ct * 16 + col];
  dst[i] = (_Float16)v;
}

__global__ void embed_kernel(const int* __restrict__ types,
                             const float* __restrict__ emb,
                             float* __restrict__ x) {
  int i = blockIdx.x * blockDim.x + threadIdx.x;
  if (i >= NATOM * HD) return;
  int a = i >> 7, c = i & (HD - 1);
  x[i] = emb[types[a] * HD + c];
}

__global__ void zero_kernel(float* __restrict__ p, int n) {
  int i = blockIdx.x * blockDim.x + threadIdx.x;
  if (i < n) p[i] = 0.0f;
}

// h = x @ lin1 (no bias), h stored f16.  One 16-row tile per wave.
__global__ void node_lin1_kernel(const float* __restrict__ x,
                                 const _Float16* __restrict__ wp,
                                 _Float16* __restrict__ h16) {
  const int NT = NATOM / 16;  // 2500
  int wv = blockIdx.x * 8 + (threadIdx.x >> 5);
  if (wv >= NT) return;                       // wave-uniform
  int lane = threadIdx.x & 31;
  int hs = lane >> 4, col = lane & 15;
  const float* xrow = x + (size_t)(wv * 16 + (lane & 15)) * HD;

  v16h a[4];
#pragma unroll
  for (int kc = 0; kc < 4; ++kc) a[kc] = load_a_f32(xrow, kc * 32, lane);

#pragma unroll
  for (int ct = 0; ct < 8; ++ct) {
    v8f acc = splat8(0.0f);
#pragma unroll
    for (int kc = 0; kc < 4; ++kc)
      acc = wmma16(a[kc], load_b_packed(wp, kc, ct, lane), acc);
#pragma unroll
    for (int g = 0; g < 8; ++g)
      h16[(size_t)(wv * 16 + g + hs * 8) * HD + ct * 16 + col] = (_Float16)acc[g];
  }
}

// fused edge pipeline: rbf -> GEMM1 -> tanh -> GEMM2 -> cutoff -> *h[src] -> atomic agg[dst]
__global__ void edge_kernel(const float* __restrict__ ew,
                            const int* __restrict__ srcI,
                            const int* __restrict__ dstI,
                            const _Float16* __restrict__ fw1_p,
                            const float* __restrict__ fb1,
                            const _Float16* __restrict__ fw2_p,
                            const float* __restrict__ fb2,
                            const _Float16* __restrict__ h16,
                            float* __restrict__ agg) {
  __shared__ _Float16 sTile[8][16 * HD];   // per-wave tanh staging (4KB each)
  __shared__ float    sC[8][16];
  __shared__ int      sS[8][16];
  __shared__ int      sD[8][16];

  const float PI = 3.14159265358979323846f;
  const float delta = CUTF / (float)(RB - 1);
  const float coeff = -0.5f / (delta * delta);

  int wv = threadIdx.x >> 5;
  int lane = threadIdx.x & 31;
  int hs = lane >> 4, col = lane & 15;
  int tile = blockIdx.x * 8 + wv;          // 40000 tiles exactly
  int ebase = tile * 16;

  if (lane < 16) {
    int e = ebase + lane;
    float d = ew[e];
    float c = 0.5f * (__cosf(PI * d / CUTF) + 1.0f);
    sC[wv][lane] = (d < CUTF) ? c : 0.0f;
    sS[wv][lane] = srcI[e];
    sD[wv][lane] = dstI[e];
  }

  // rbf directly in A-fragment layout: element i<8 -> K = kc*32+8*hs+i,
  // element i>=8 -> K = kc*32+16+8*hs+(i-8)
  float dme = ew[ebase + (lane & 15)];
  v16h ra[2];
#pragma unroll
  for (int kc = 0; kc < 2; ++kc)
#pragma unroll
    for (int i = 0; i < 16; ++i) {
      int k = kc * 32 + hs * 8 + ((i < 8) ? i : (i + 8));
      float t = dme - (float)k * delta;
      ra[kc][i] = (_Float16)fast_exp(coeff * t * t);
    }

  // GEMM1: (16 x 64) @ fw1 (64 x 128) + fb1, tanh -> LDS tile
#pragma unroll
  for (int ct = 0; ct < 8; ++ct) {
    v8f acc = splat8(fb1[ct * 16 + col]);
    acc = wmma16(ra[0], load_b_packed(fw1_p, 0, ct, lane), acc);
    acc = wmma16(ra[1], load_b_packed(fw1_p, 1, ct, lane), acc);
#pragma unroll
    for (int g = 0; g < 8; ++g)
      sTile[wv][(g + hs * 8) * HD + ct * 16 + col] = (_Float16)fast_tanh(acc[g]);
  }
  asm volatile("s_wait_dscnt 0" ::: "memory");

  v16h a2[4];
#pragma unroll
  for (int kc = 0; kc < 4; ++kc) a2[kc] = load_a_lds(&sTile[wv][0], kc * 32, lane);

  // GEMM2: (16 x 128) @ fw2 (128 x 128) + fb2; then scatter messages
#pragma unroll
  for (int ct = 0; ct < 8; ++ct) {
    v8f acc = splat8(fb2[ct * 16 + col]);
#pragma unroll
    for (int kc = 0; kc < 4; ++kc)
      acc = wmma16(a2[kc], load_b_packed(fw2_p, kc, ct, lane), acc);
#pragma unroll
    for (int g = 0; g < 8; ++g) {
      int er = g + hs * 8;                     // edge row within tile
      float wval = acc[g] * sC[wv][er];
      int s = sS[wv][er];
      int dn = sD[wv][er];
      float hv = (float)h16[(size_t)s * HD + ct * 16 + col];
      unsafeAtomicAdd(&agg[(size_t)dn * HD + ct * 16 + col], wval * hv);
    }
  }
}

// y = tanh(agg@lin2 + b2) @ lin_w + lin_b;  x += y
__global__ void node_update_kernel(const float* __restrict__ agg,
                                   const _Float16* __restrict__ lin2_p,
                                   const float* __restrict__ lin2_b,
                                   const _Float16* __restrict__ linw_p,
                                   const float* __restrict__ lin_b,
                                   float* __restrict__ x) {
  __shared__ _Float16 sTile[8][16 * HD];
  const int NT = NATOM / 16;
  int wv = blockIdx.x * 8 + (threadIdx.x >> 5);
  if (wv >= NT) return;                       // wave-uniform
  int wloc = threadIdx.x >> 5;
  int lane = threadIdx.x & 31;
  int hs = lane >> 4, col = lane & 15;
  int tbase = wv * 16;

  // stage agg tile as f16 into LDS (vectorized: float4 in, 4 halves out)
#pragma unroll
  for (int i = 0; i < 16; ++i) {
    int ofs = lane * 64 + i * 4;
    float4 v = *(const float4*)(agg + (size_t)tbase * HD + ofs);
    v4h hv;
    hv[0] = (_Float16)v.x; hv[1] = (_Float16)v.y;
    hv[2] = (_Float16)v.z; hv[3] = (_Float16)v.w;
    *(v4h*)&sTile[wloc][ofs] = hv;
  }
  asm volatile("s_wait_dscnt 0" ::: "memory");

  v16h a[4];
#pragma unroll
  for (int kc = 0; kc < 4; ++kc) a[kc] = load_a_lds(&sTile[wloc][0], kc * 32, lane);

  // GEMM1 + tanh -> LDS (overwrite; A already in registers)
#pragma unroll
  for (int ct = 0; ct < 8; ++ct) {
    v8f acc = splat8(lin2_b[ct * 16 + col]);
#pragma unroll
    for (int kc = 0; kc < 4; ++kc)
      acc = wmma16(a[kc], load_b_packed(lin2_p, kc, ct, lane), acc);
#pragma unroll
    for (int g = 0; g < 8; ++g)
      sTile[wloc][(g + hs * 8) * HD + ct * 16 + col] = (_Float16)fast_tanh(acc[g]);
  }
  asm volatile("s_wait_dscnt 0" ::: "memory");

  v16h a2[4];
#pragma unroll
  for (int kc = 0; kc < 4; ++kc) a2[kc] = load_a_lds(&sTile[wloc][0], kc * 32, lane);

#pragma unroll
  for (int ct = 0; ct < 8; ++ct) {
    v8f acc = splat8(lin_b[ct * 16 + col]);
#pragma unroll
    for (int kc = 0; kc < 4; ++kc)
      acc = wmma16(a2[kc], load_b_packed(linw_p, kc, ct, lane), acc);
#pragma unroll
    for (int g = 0; g < 8; ++g)
      x[(size_t)(tbase + g + hs * 8) * HD + ct * 16 + col] += acc[g];   // residual
  }
}

// energy = tanh(x @ o1_w + o1_b) @ o2_w + o2_b    (H -> 64 -> 1)
__global__ void output_kernel(const float* __restrict__ x,
                              const float* __restrict__ o1w,
                              const float* __restrict__ o1b,
                              const float* __restrict__ o2w,
                              const float* __restrict__ o2b,
                              float* __restrict__ out) {
  __shared__ float red[64];
  int i = blockIdx.x;          // atom
  int j = threadIdx.x;         // hidden unit 0..63
  float s = o1b[j];
#pragma unroll 8
  for (int c = 0; c < HD; ++c) s += x[(size_t)i * HD + c] * o1w[c * 64 + j];
  red[j] = fast_tanh(s) * o2w[j];
  __syncthreads();
#pragma unroll
  for (int st = 32; st > 0; st >>= 1) {
    if (j < st) red[j] += red[j + st];
    __syncthreads();
  }
  if (j == 0) out[i] = red[0] + o2b[0];
}

// ---------------------------------------------------------------- launch

extern "C" void kernel_launch(void* const* d_in, const int* in_sizes, int n_in,
                              void* d_out, int out_size, void* d_ws, size_t ws_size,
                              hipStream_t stream) {
  const int*   atom_types  = (const int*)d_in[0];
  const int*   edge_index  = (const int*)d_in[1];
  const float* edge_weight = (const float*)d_in[2];
  const float* emb         = (const float*)d_in[3];
  const float* lin1_w      = (const float*)d_in[4];
  const float* fw1         = (const float*)d_in[5];
  const float* fb1         = (const float*)d_in[6];
  const float* fw2         = (const float*)d_in[7];
  const float* fb2         = (const float*)d_in[8];
  const float* lin2_w      = (const float*)d_in[9];
  const float* lin2_b      = (const float*)d_in[10];
  const float* lin_w       = (const float*)d_in[11];
  const float* lin_b       = (const float*)d_in[12];
  const float* o1_w        = (const float*)d_in[13];
  const float* o1_b        = (const float*)d_in[14];
  const float* o2_w        = (const float*)d_in[15];
  const float* o2_b        = (const float*)d_in[16];
  float* out = (float*)d_out;

  char* ws = (char*)d_ws;
  size_t off = 0;
  float* x   = (float*)(ws + off);    off += (size_t)NATOM * HD * 4;   // 20.48 MB
  float* agg = (float*)(ws + off);    off += (size_t)NATOM * HD * 4;   // 20.48 MB
  _Float16* h16    = (_Float16*)(ws + off); off += (size_t)NATOM * HD * 2;
  _Float16* p_lin1 = (_Float16*)(ws + off); off += (size_t)NBLK * HD * HD * 2;
  _Float16* p_fw1  = (_Float16*)(ws + off); off += (size_t)NBLK * RB * HD * 2;
  _Float16* p_fw2  = (_Float16*)(ws + off); off += (size_t)NBLK * HD * HD * 2;
  _Float16* p_lin2 = (_Float16*)(ws + off); off += (size_t)NBLK * HD * HD * 2;
  _Float16* p_linw = (_Float16*)(ws + off); off += (size_t)NBLK * HD * HD * 2;

  const int* srcI = edge_index;          // edge_index[0, :]
  const int* dstI = edge_index + NEDGE;  // edge_index[1, :]

  // --- pack weights into B-fragment-major f16 (tiny one-shot work) ---
  int nHH = NBLK * HD * HD, nRH = NBLK * RB * HD;
  pack_b_kernel<<<(nHH + 255) / 256, 256, 0, stream>>>(lin1_w, p_lin1, 4, nHH);
  pack_b_kernel<<<(nRH + 255) / 256, 256, 0, stream>>>(fw1,    p_fw1,  2, nRH);
  pack_b_kernel<<<(nHH + 255) / 256, 256, 0, stream>>>(fw2,    p_fw2,  4, nHH);
  pack_b_kernel<<<(nHH + 255) / 256, 256, 0, stream>>>(lin2_w, p_lin2, 4, nHH);
  pack_b_kernel<<<(nHH + 255) / 256, 256, 0, stream>>>(lin_w,  p_linw, 4, nHH);

  // --- embedding ---
  int nNH = NATOM * HD;
  embed_kernel<<<(nNH + 255) / 256, 256, 0, stream>>>(atom_types, emb, x);

  const int nodeBlocks = (NATOM / 16 + 7) / 8;   // 313
  const int edgeBlocks = NEDGE / 16 / 8;         // 5000

  for (int b = 0; b < NBLK; ++b) {
    zero_kernel<<<(nNH + 255) / 256, 256, 0, stream>>>(agg, nNH);
    node_lin1_kernel<<<nodeBlocks, 256, 0, stream>>>(x, p_lin1 + (size_t)b * HD * HD, h16);
    edge_kernel<<<edgeBlocks, 256, 0, stream>>>(
        edge_weight, srcI, dstI,
        p_fw1 + (size_t)b * RB * HD, fb1 + (size_t)b * HD,
        p_fw2 + (size_t)b * HD * HD, fb2 + (size_t)b * HD,
        h16, agg);
    node_update_kernel<<<nodeBlocks, 256, 0, stream>>>(
        agg,
        p_lin2 + (size_t)b * HD * HD, lin2_b + (size_t)b * HD,
        p_linw + (size_t)b * HD * HD, lin_b + (size_t)b * HD,
        x);
  }

  output_kernel<<<NATOM, 64, 0, stream>>>(x, o1_w, o1_b, o2_w, o2_b, out);
}
